// GNN_6227702579880
// MI455X (gfx1250) — matrix operand order repaired and checked by reading, not compiled, not run on hardware
//
#include <hip/hip_runtime.h>
#include <hip/hip_bf16.h>

typedef __attribute__((ext_vector_type(16))) _Float16     v16h;
typedef __attribute__((ext_vector_type(8)))  _Float16     v8h;
typedef __attribute__((ext_vector_type(8)))  float        v8f;
typedef __attribute__((ext_vector_type(4)))  unsigned int u32x4;
typedef __attribute__((ext_vector_type(8)))  int          i32x8;
typedef __attribute__((ext_vector_type(4)))  int          i32x4;

#define B_  16
#define N_  96
#define D_  128
#define NC  10
#define PAD_LD 136   // 128 + 8 halves: 272B row stride -> conflict-free, 16B aligned

// ---------------------------------------------------------------------------
// WMMA helpers (CDNA5 16x16x32 f16 -> f32 accumulate)
// A-fragment: 16x32 (MxK) f16, row-major source with stride ld (halves).
//   lanes 0-15 : M=lane,      halves 0-7 -> K=k0..k0+7,    halves 8-15 -> K=k0+16..k0+23
//   lanes 16-31: M=lane-16,   halves 0-7 -> K=k0+8..k0+15, halves 8-15 -> K=k0+24..k0+31
__device__ inline v16h loadA16(const _Float16* base, int ld, int row0, int k0, int lane) {
    const _Float16* p = base + (row0 + (lane & 15)) * ld + k0 + ((lane & 16) ? 8 : 0);
    v8h lo = *(const v8h*)p;
    v8h hi = *(const v8h*)(p + 16);
    v16h a;
#pragma unroll
    for (int i = 0; i < 8; ++i) { a[i] = lo[i]; a[i + 8] = hi[i]; }
    return a;
}

// B-fragment: 32x16 (KxN). Source is B^T row-major [n][k] with stride ld: each lane
// reads K halves in the same order as the A-fragment (two b128 loads).
__device__ inline v16h loadBT16(const _Float16* base, int ld, int n0, int k0, int lane) {
    const _Float16* p = base + (n0 + (lane & 15)) * ld + k0 + ((lane & 16) ? 8 : 0);
    v8h lo = *(const v8h*)p;
    v8h hi = *(const v8h*)(p + 16);
    v16h b;
#pragma unroll
    for (int i = 0; i < 8; ++i) { b[i] = lo[i]; b[i + 8] = hi[i]; }
    return b;
}

__device__ inline v8f wmma16(v16h a, v16h b, v8f c) {
    return __builtin_amdgcn_wmma_f32_16x16x32_f16(false, a, false, b, (short)0, c,
                                                  false, false);
}

// ---------------------------------------------------------------------------
// K0: weights -> f16, transposed ([out][in]) for contiguous B-fragment loads
__global__ void k_prep(const float* __restrict__ W1, const float* __restrict__ W2,
                       const float* __restrict__ Wo1,
                       _Float16* __restrict__ W1eT, _Float16* __restrict__ W1xT,
                       _Float16* __restrict__ W2T,  _Float16* __restrict__ Wo1T) {
    int idx = blockIdx.x * blockDim.x + threadIdx.x;
    for (int i = idx; i < D_ * D_; i += gridDim.x * blockDim.x) {
        int e = i >> 7, d = i & 127;                 // i = e*128 + d
        W1eT[i] = (_Float16)W1[d * D_ + e];          // W1e = W1[:128]
        W1xT[i] = (_Float16)W1[(D_ + d) * D_ + e];   // W1x = W1[128:]
        W2T[i]  = (_Float16)W2[d * D_ + e];
        Wo1T[i] = (_Float16)Wo1[d * D_ + e];
    }
}

// K1: x = embed_W[tok]
__global__ void k_embed(const int* __restrict__ tok, const float* __restrict__ embed_W,
                        float* __restrict__ x) {
    int row = blockIdx.x;
    int t = tok[row];
    x[row * D_ + threadIdx.x] = embed_W[t * D_ + threadIdx.x];
}

// K2: Y(M,128) = X(M,128) @ W, W supplied transposed f16. 16 M-rows per workgroup.
__global__ void k_proj(const float* __restrict__ X, const _Float16* __restrict__ WT,
                       float* __restrict__ Y) {
    __shared__ __align__(16) _Float16 Xh[16 * PAD_LD];
    int tid = threadIdx.x, lane = tid & 31, wave = tid >> 5;
    int mbase = blockIdx.x * 16;
    for (int i = tid; i < 16 * D_; i += 256) {
        int r = i >> 7, d = i & 127;
        Xh[r * PAD_LD + d] = (_Float16)X[(mbase + r) * D_ + d];
    }
    __syncthreads();
    int nt = wave;                                   // 8 waves cover N=128
    v8f acc = {};
#pragma unroll
    for (int kt = 0; kt < 4; ++kt) {
        v16h a = loadA16(Xh, PAD_LD, 0, kt * 32, lane);
        v16h b = loadBT16(WT, D_, nt * 16, kt * 32, lane);
        acc = wmma16(a, b, acc);
    }
    int n = nt * 16 + (lane & 15);
    int mofs = (lane & 16) ? 8 : 0;
#pragma unroll
    for (int v = 0; v < 8; ++v)
        Y[(mbase + mofs + v) * D_ + n] = acc[v];
}

// K3: per (b,i): h[j,d] = relu(e_proj[i,d] + xj_proj[j,d] + b1[d]) built in LDS (f16),
//     m = relu(h @ W2 + b2) via WMMA, agg[d] = sum_j A[b,j,i]*m[j,d] folded out of the
//     accumulators (register partials + shfl_xor), x_mid = x + agg.
// Dynamic LDS layout (16B aligned):
//   [0)       W2s : 128 x PAD_LD f16  = 34816 B   (filled by TDM w/ pad, or fallback)
//   [34816)   Hh  :  96 x PAD_LD f16  = 26112 B
//   [60928)   eprojs : 128 f32
//   [61440)   b1s    : 128 f32
//   [61952)   acol   :  96 f32        -> total 62336 B
#define K3_LDS_BYTES 62336
__global__ void k_pairwise(const float* __restrict__ A, const float* __restrict__ eproj,
                           const float* __restrict__ xjproj, const float* __restrict__ x,
                           const _Float16* __restrict__ W2T, const float* __restrict__ b1,
                           const float* __restrict__ b2, float* __restrict__ x_mid) {
    extern __shared__ __align__(16) char smem[];
    _Float16* W2s    = (_Float16*)smem;
    _Float16* Hh     = (_Float16*)(smem + 34816);
    float*    eprojs = (float*)(smem + 60928);
    float*    b1s    = (float*)(smem + 61440);
    float*    acol   = (float*)(smem + 61952);

    int tid = threadIdx.x, lane = tid & 31, wave = tid >> 5;
    int b = blockIdx.x / N_, i = blockIdx.x % N_;
    int rowbase = b * N_;

#if __has_builtin(__builtin_amdgcn_tensor_load_to_lds) && __has_builtin(__builtin_amdgcn_s_wait_tensorcnt)
    if (wave == 0) {
        // TDM: 2D tensor 128x128 f16, full-tile load, LDS pad 4 DWORDs (16B) after
        // every 64 DWORDs (256B = one row) -> padded stride PAD_LD in LDS.
        unsigned lds_base = __builtin_amdgcn_groupstaticsize();   // dyn-shared base
        unsigned long long ga = (unsigned long long)(const void*)W2T;
        u32x4 g0 = { 1u,                                          // count=1, user mode
                     lds_base,                                    // lds_addr
                     (unsigned)(ga & 0xffffffffu),                // global_addr[31:0]
                     (unsigned)(((ga >> 32) & 0x01ffffffu)        // global_addr[56:32]
                                | 0x80000000u) };                 // type=2 ("image")
        i32x8 g1 = { (int)((1u << 16)      // data_size = 2B
                           | (1u << 20)    // pad_enable
                           | (5u << 22)    // pad_interval: 64 DWORDs
                           | (3u << 25)),  // pad_amount:   4 DWORDs
                     (int)(128u << 16),    // tensor_dim0[15:0]  = 128 (bits 79:48)
                     (int)(128u << 16),    // dim0 hi=0 | tensor_dim1[15:0] = 128
                     (int)(128u << 16),    // dim1 hi=0 | tile_dim0 = 128
                     (int)128u,            // tile_dim1 = 128, tile_dim2 = 0
                     (int)128u,            // tensor_dim0_stride[31:0] = 128
                     0, 0 };
        i32x4 gz = { 0, 0, 0, 0 };
#if defined(__clang_major__) && __clang_major__ >= 23
        i32x8 gz8 = { 0, 0, 0, 0, 0, 0, 0, 0 };
        __builtin_amdgcn_tensor_load_to_lds(g0, g1, gz, gz, gz8, 0);
#else
        __builtin_amdgcn_tensor_load_to_lds(g0, g1, gz, gz, 0);
#endif
        __builtin_amdgcn_s_wait_tensorcnt((short)0);
    }
#else
    for (int k = tid; k < 2048; k += 256) {          // 2048 uint4 = 32KB payload
        int row = k >> 4, c = k & 15;
        ((uint4*)(W2s + row * PAD_LD))[c] = ((const uint4*)W2T)[k];
    }
#endif
    for (int d = tid; d < D_; d += 256) {
        eprojs[d] = eproj[(rowbase + i) * D_ + d];
        b1s[d] = b1[d];
    }
    for (int j = tid; j < N_; j += 256) acol[j] = A[(rowbase + j) * N_ + i];
    __syncthreads();

    for (int idx = tid; idx < N_ * D_; idx += 256) {
        int j = idx >> 7, d = idx & 127;
        float v = eprojs[d] + xjproj[(rowbase + j) * D_ + d] + b1s[d];
        Hh[j * PAD_LD + d] = (_Float16)fmaxf(v, 0.f);
    }
    __syncthreads();

    int nt = wave;                                   // wave exclusively owns 16 columns
    v16h bfr[4];
#pragma unroll
    for (int kt = 0; kt < 4; ++kt) bfr[kt] = loadBT16(W2s, PAD_LD, nt * 16, kt * 32, lane);

    int n = nt * 16 + (lane & 15);
    float b2v = b2[n];
    int jofs = (lane & 16) ? 8 : 0;
    float pacc = 0.f;

    for (int jt = 0; jt < 6; ++jt) {                 // 96 j's in 6 tiles
        v8f acc = {};
#pragma unroll
        for (int kt = 0; kt < 4; ++kt) {
            v16h a = loadA16(Hh, PAD_LD, jt * 16, kt * 32, lane);
            acc = wmma16(a, bfr[kt], acc);
        }
#pragma unroll
        for (int v = 0; v < 8; ++v) {                // row M = jt*16 + jofs + v
            float m = fmaxf(acc[v] + b2v, 0.f);
            pacc += acol[jt * 16 + jofs + v] * m;
        }
    }
    pacc += __shfl_xor(pacc, 16, 32);                // combine the two j-halves
    if (lane < 16) {
        int o = (rowbase + i) * D_ + n;
        x_mid[o] = x[o] + pacc;
    }
}

// ---------------------------------------------------------------------------
// Shared head GEMV: t = relu(xrow @ Wo1 + bo1), vectorized v8h weight loads.
__device__ inline void head_hidden(const float* xrow, const _Float16* __restrict__ Wo1T,
                                   const float* __restrict__ bo1, float* trow, int lane) {
#pragma unroll
    for (int q = 0; q < 4; ++q) {
        int e = lane + 32 * q;
        float s = bo1[e];
        const v8h* wr = (const v8h*)(Wo1T + e * D_);
#pragma unroll 4
        for (int c8 = 0; c8 < 16; ++c8) {
            v8h w = wr[c8];
#pragma unroll
            for (int u = 0; u < 8; ++u) s += xrow[c8 * 8 + u] * (float)w[u];
        }
        trow[e] = fmaxf(s, 0.f);
    }
}

// K4: x = x_mid + softmax(head(x_mid)/tau) @ centroids   (one wave per row)
__global__ void k_discretize(const float* __restrict__ x_mid,
                             const _Float16* __restrict__ Wo1T,
                             const float* __restrict__ bo1, const float* __restrict__ Wo2,
                             const float* __restrict__ embed_W, const int* __restrict__ tau_p,
                             float* __restrict__ x) {
    __shared__ float xrow[8][D_], trow[8][D_], lg[8][16], av[8][16];
    int tid = threadIdx.x, lane = tid & 31, wave = tid >> 5;
    int row = blockIdx.x * 8 + wave;

    for (int d = lane; d < D_; d += 32) xrow[wave][d] = x_mid[row * D_ + d];
    __syncthreads();
    head_hidden(xrow[wave], Wo1T, bo1, trow[wave], lane);
    __syncthreads();
    if (lane < NC) {
        float s = 0.f;
        for (int d = 0; d < D_; ++d) s += trow[wave][d] * Wo2[d * NC + lane];
        lg[wave][lane] = s;
    }
    __syncthreads();
    if (lane == 0) {
        float tau = (float)(*tau_p);
        float mx = -3.4e38f;
        for (int c = 0; c < NC; ++c) mx = fmaxf(mx, lg[wave][c] / tau);
        float sum = 0.f;
        for (int c = 0; c < NC; ++c) {
            float ev = __expf(lg[wave][c] / tau - mx);
            av[wave][c] = ev; sum += ev;
        }
        float inv = 1.f / sum;
        for (int c = 0; c < NC; ++c) av[wave][c] *= inv;
    }
    __syncthreads();
    const float* cent = embed_W + 4 * D_;            // centroids = embed_W[4:]
#pragma unroll
    for (int q = 0; q < 4; ++q) {
        int d = lane + 32 * q;
        float s = 0.f;
        for (int c = 0; c < NC; ++c) s += av[wave][c] * cent[c * D_ + d];
        x[row * D_ + d] = xrow[wave][d] + s;
    }
}

// K5: x_all = head(x) for all rows; rows with n==0 also feed `out`
__global__ void k_head(const float* __restrict__ xf, const _Float16* __restrict__ Wo1T,
                       const float* __restrict__ bo1, const float* __restrict__ Wo2,
                       float* __restrict__ out) {
    __shared__ float xrow[8][D_], trow[8][D_];
    int tid = threadIdx.x, lane = tid & 31, wave = tid >> 5;
    int row = blockIdx.x * 8 + wave;

    for (int d = lane; d < D_; d += 32) xrow[wave][d] = xf[row * D_ + d];
    __syncthreads();
    head_hidden(xrow[wave], Wo1T, bo1, trow[wave], lane);
    __syncthreads();
    if (lane < NC) {
        float s = 0.f;
        for (int d = 0; d < D_; ++d) s += trow[wave][d] * Wo2[d * NC + lane];
        out[B_ * NC + row * NC + lane] = s;                   // x_all
        if ((row % N_) == 0) out[(row / N_) * NC + lane] = s; // out = x_all[:,0,:]
    }
}

// ---------------------------------------------------------------------------
extern "C" void kernel_launch(void* const* d_in, const int* in_sizes, int n_in,
                              void* d_out, int out_size, void* d_ws, size_t ws_size,
                              hipStream_t stream) {
    (void)in_sizes; (void)n_in; (void)out_size; (void)ws_size;
    const int*   tok     = (const int*)d_in[0];
    const float* A       = (const float*)d_in[1];
    const float* embed_W = (const float*)d_in[2];
    const float* W1      = (const float*)d_in[3];
    const float* b1      = (const float*)d_in[4];
    const float* W2      = (const float*)d_in[5];
    const float* b2      = (const float*)d_in[6];
    const float* Wo1     = (const float*)d_in[7];
    const float* bo1     = (const float*)d_in[8];
    const float* Wo2     = (const float*)d_in[9];
    const int*   tau_p   = (const int*)d_in[11];
    float* out = (float*)d_out;

    char* ws = (char*)d_ws;
    size_t off = 0;
    auto alloc = [&](size_t bytes) {
        void* p = ws + off;
        off += (bytes + 255) & ~size_t(255);
        return p;
    };
    float*    x      = (float*)alloc(size_t(B_) * N_ * D_ * 4);
    float*    eproj  = (float*)alloc(size_t(B_) * N_ * D_ * 4);
    float*    xjproj = (float*)alloc(size_t(B_) * N_ * D_ * 4);
    float*    x_mid  = (float*)alloc(size_t(B_) * N_ * D_ * 4);
    _Float16* W1eT   = (_Float16*)alloc(size_t(D_) * D_ * 2);
    _Float16* W1xT   = (_Float16*)alloc(size_t(D_) * D_ * 2);
    _Float16* W2T    = (_Float16*)alloc(size_t(D_) * D_ * 2);
    _Float16* Wo1T   = (_Float16*)alloc(size_t(D_) * D_ * 2);

    k_prep<<<64, 256, 0, stream>>>(W1, W2, Wo1, W1eT, W1xT, W2T, Wo1T);
    k_embed<<<B_ * N_, 128, 0, stream>>>(tok, embed_W, x);
    k_proj<<<(B_ * N_) / 16, 256, 0, stream>>>(x, W1eT, eproj);

    for (int r = 0; r < 5; ++r) {
        k_proj<<<(B_ * N_) / 16, 256, 0, stream>>>(x, W1xT, xjproj);
        k_pairwise<<<B_ * N_, 256, K3_LDS_BYTES, stream>>>(A, eproj, xjproj, x, W2T,
                                                           b1, b2, x_mid);
        if (r < 4)
            k_discretize<<<192, 256, 0, stream>>>(x_mid, Wo1T, bo1, Wo2, embed_W, tau_p, x);
    }
    k_head<<<192, 256, 0, stream>>>(x_mid, Wo1T, bo1, Wo2, out);
}
